// GraphormerAttentionHead_76433237999808
// MI455X (gfx1250) — matrix operand order repaired and checked by reading, not compiled, not run on hardware
//
#include <hip/hip_runtime.h>
#include <hip/hip_bf16.h>
#include <math.h>

// ---------------------------------------------------------------------------
// Graphormer attention head for MI455X (gfx1250, wave32, WMMA).
//
// Pipeline (all on `stream`):
//  K1 convert     : x -> xh (f16), [Wq|Wk|Wv]^T -> WhT (f16), zero CSR counters
//  K2 qkv (WMMA)  : qh = (x@Wq+bq)/8 (f16), kh = x@Wk+bk (f16), vhT = (x@Wv+bv)^T (f16)
//  K3 count       : rowcnt[src[p]]++
//  K4 scan        : rowptr = exclusive prefix of rowcnt
//  K5 dots+fill   : per-pair path-edge dot, scatter into CSR (colidx/val)
//  K6 ablk (WMMA) : per graph, A_blk = q_blk @ k_blk^T   (q already has 1/8)
//  K7 rowstats    : per row, stream full b row (non-temporal; b > L2);
//                   online softmax (max,sumexp), exact sparse-c via bitmap+CSR
//  K8 attnv(WMMA) : per graph, attn_blk = exp(l-M)/S (f16 in LDS), out = attn@v
// ---------------------------------------------------------------------------

typedef _Float16 v16h __attribute__((ext_vector_type(16)));
typedef _Float16 v8h  __attribute__((ext_vector_type(8)));
typedef float    v8f  __attribute__((ext_vector_type(8)));
typedef float    v4f  __attribute__((ext_vector_type(4)));

#define DIM_IN   256
#define DIM_Q    64
#define QKV_COLS 192          // q|k|v concatenated
#define NCT      12           // 192/16 column tiles in QKV GEMM
#define MAXPATH  5
#define NEGF     (-1000000.0f)

// ---- WMMA fragment helpers (ISA 7.12.2 layouts, wave32) -------------------
// A: 16x32 f16, lane m=lane&15 holds row m; hi=lane>>4 selects K-halves:
//    elems 0..7 = K in [k0+hi*8, +8), elems 8..15 = K in [k0+16+hi*8, +8)
__device__ __forceinline__ v16h load_a_frag(const _Float16* rowbase, int hi, int k0) {
    const _Float16* p = rowbase + k0 + hi * 8;
    v8h lo = *(const v8h*)(p);
    v8h hh = *(const v8h*)(p + 16);
    return __builtin_shufflevector(lo, hh, 0,1,2,3,4,5,6,7,8,9,10,11,12,13,14,15);
}
// B: 32x16 f16, lane n=lane&15 holds column n; same K order as A:
//    elems 0..7 = K in [k0+hi*8, +8), elems 8..15 = K in [k0+16+hi*8, +8)
// (caller passes a pointer where K runs contiguously)
__device__ __forceinline__ v16h load_b_frag(const _Float16* kcontig, int hi, int k0) {
    const _Float16* p = kcontig + k0 + hi * 8;
    v8h lo = *(const v8h*)(p);
    v8h hh = *(const v8h*)(p + 16);
    return __builtin_shufflevector(lo, hh, 0,1,2,3,4,5,6,7,8,9,10,11,12,13,14,15);
}

__device__ __forceinline__ void softmax_w(const float* attn_raw, float& w0, float& w1) {
    float e0 = __expf(attn_raw[0]);
    float e1 = __expf(attn_raw[1]);
    float inv = 1.0f / (e0 + e1);
    w0 = e0 * inv; w1 = e1 * inv;
}

// Branch-free online (max, sumexp) update: avoids wave32 divergence in the
// hot streaming loop; the extra v_exp_f32 is a TRANS op that co-executes.
__device__ __forceinline__ void online_update(float& m, float& s, float x) {
    float mn = fmaxf(m, x);
    s = s * __expf(m - mn) + __expf(x - mn);
    m = mn;
}

// ---------------------------------------------------------------------------
// K1: conversions + zero CSR counters
__global__ void k1_convert(const float* __restrict__ x,
                           const float* __restrict__ Wq,
                           const float* __restrict__ Wk,
                           const float* __restrict__ Wv,
                           _Float16* __restrict__ xh,
                           _Float16* __restrict__ WhT,
                           int* __restrict__ rowcnt,
                           int* __restrict__ rowfill,
                           int N) {
    int stride = gridDim.x * blockDim.x;
    int tid = blockIdx.x * blockDim.x + threadIdx.x;
    int totalx = N * DIM_IN;
    for (int i = tid; i < totalx; i += stride) xh[i] = (_Float16)x[i];
    for (int i = tid; i < QKV_COLS * DIM_IN; i += stride) {
        int oc = i >> 8, ic = i & 255;                 // WhT[oc][ic] = W[ic][oc%64]
        const float* W = (oc < DIM_Q) ? Wq : ((oc < 2 * DIM_Q) ? Wk : Wv);
        WhT[i] = (_Float16)W[ic * DIM_Q + (oc & (DIM_Q - 1))];
    }
    for (int i = tid; i < 2 * N; i += stride) {
        if (i < N) rowcnt[i] = 0; else rowfill[i - N] = 0;
    }
}

// ---------------------------------------------------------------------------
// K2: QKV projection, one wave per 16x16 output tile, K=256 in 8 WMMA steps
__global__ void k2_qkv(const _Float16* __restrict__ xh,
                       const _Float16* __restrict__ WhT,
                       const float* __restrict__ bq,
                       const float* __restrict__ bk,
                       const float* __restrict__ bv,
                       _Float16* __restrict__ qh,
                       _Float16* __restrict__ kh,
                       _Float16* __restrict__ vhT,
                       int N, int nTiles) {
    int wave = threadIdx.x >> 5;
    int lane = threadIdx.x & 31;
    int tile = blockIdx.x * 8 + wave;
    if (tile >= nTiles) return;                        // wave-uniform
    int ct = tile % NCT, rt = tile / NCT;
    int row0 = rt * 16, col0 = ct * 16;
    int m = lane & 15, hi = lane >> 4;

    const _Float16* arow = xh  + (size_t)(row0 + m) * DIM_IN;
    const _Float16* bcol = WhT + (size_t)(col0 + m) * DIM_IN;  // n == m (lane&15)
    v8f acc = {};
    #pragma unroll
    for (int k0 = 0; k0 < DIM_IN; k0 += 32) {
        v16h a = load_a_frag(arow, hi, k0);
        v16h bfr = load_b_frag(bcol, hi, k0);
        acc = __builtin_amdgcn_wmma_f32_16x16x32_f16(false, a, false, bfr,
                                                     (short)0, acc, false, false);
    }
    int col = col0 + m;
    if (col < DIM_Q) {                                  // q region, fold 1/sqrt(64)
        float bias = bq[col];
        #pragma unroll
        for (int r = 0; r < 8; ++r) {
            int row = row0 + hi * 8 + r;
            qh[(size_t)row * DIM_Q + col] = (_Float16)((acc[r] + bias) * 0.125f);
        }
    } else if (col < 2 * DIM_Q) {                       // k region
        float bias = bk[col - DIM_Q];
        #pragma unroll
        for (int r = 0; r < 8; ++r) {
            int row = row0 + hi * 8 + r;
            kh[(size_t)row * DIM_Q + (col - DIM_Q)] = (_Float16)(acc[r] + bias);
        }
    } else {                                            // v region, stored transposed
        float bias = bv[col - 2 * DIM_Q];
        _Float16* p = vhT + (size_t)(col - 2 * DIM_Q) * N + row0 + hi * 8;
        #pragma unroll
        for (int r = 0; r < 8; ++r) p[r] = (_Float16)(acc[r] + bias);
    }
}

// ---------------------------------------------------------------------------
// K3: CSR counting
__global__ void k3_count(const int* __restrict__ src, int* __restrict__ rowcnt, int P) {
    int p = blockIdx.x * blockDim.x + threadIdx.x;
    if (p < P) atomicAdd(&rowcnt[src[p]], 1);
}

// K4: single-block exclusive scan -> rowptr[0..N]
__global__ void k4_scan(const int* __restrict__ rowcnt, int* __restrict__ rowptr, int N) {
    __shared__ int part[256];
    int tid = threadIdx.x;
    int chunk = (N + 255) >> 8;
    int start = tid * chunk;
    int end = start + chunk; if (end > N) end = N;
    int sum = 0;
    for (int r = start; r < end; ++r) sum += rowcnt[r];
    part[tid] = sum;
    __syncthreads();
    if (tid == 0) {
        int run = 0;
        for (int t = 0; t < 256; ++t) { int v = part[t]; part[t] = run; run += v; }
    }
    __syncthreads();
    int run = part[tid];
    for (int r = start; r < end; ++r) { rowptr[r] = run; run += rowcnt[r]; }
    if (tid == 255) rowptr[N] = run;
}

// ---------------------------------------------------------------------------
// K5: per-pair path dots + CSR fill. edge_attr (4MB) is L2-resident.
__global__ void k5_dots_fill(const float* __restrict__ edge_attr,
                             const int* __restrict__ src,
                             const int* __restrict__ dst,
                             const int* __restrict__ path,
                             const float* __restrict__ edge_vector,
                             const int* __restrict__ rowptr,
                             int* __restrict__ rowfill,
                             int* __restrict__ colidx,
                             float* __restrict__ valout,
                             int P) {
    __shared__ alignas(16) float ev[MAXPATH * 64];
    for (int k = threadIdx.x; k < MAXPATH * 64; k += blockDim.x) ev[k] = edge_vector[k];
    __syncthreads();
    int p = blockIdx.x * blockDim.x + threadIdx.x;
    if (p >= P) return;
    float acc = 0.0f;
    #pragma unroll
    for (int l = 0; l < MAXPATH; ++l) {
        int e = path[p * MAXPATH + l];
        if (e >= 0) {
            const v4f* ea4 = (const v4f*)(edge_attr + (size_t)e * 64);
            const v4f* ev4 = (const v4f*)(ev + l * 64);
            float d = 0.0f;
            #pragma unroll
            for (int q = 0; q < 16; ++q) {
                v4f a = ea4[q], b = ev4[q];
                d += a[0]*b[0] + a[1]*b[1] + a[2]*b[2] + a[3]*b[3];
            }
            acc += d;
        }
    }
    acc *= (1.0f / (float)MAXPATH);                    // mean over L (incl. padding)
    int r = src[p];
    int slot = atomicAdd(&rowfill[r], 1);
    int idx = rowptr[r] + slot;
    colidx[idx] = dst[p];
    valout[idx] = acc;
}

// ---------------------------------------------------------------------------
// K6: per-graph A_blk = q_blk @ k_blk^T (64x64), 8 waves x 2 tiles, WMMA
__global__ void k6_ablk(const _Float16* __restrict__ qh,
                        const _Float16* __restrict__ kh,
                        const int* __restrict__ ptr,
                        float* __restrict__ Ablk) {
    int g = blockIdx.x;
    int blo = ptr[g];
    int wave = threadIdx.x >> 5, lane = threadIdx.x & 31;
    int m = lane & 15, hi = lane >> 4;
    #pragma unroll
    for (int t = 0; t < 2; ++t) {
        int tile = wave * 2 + t;                       // 0..15 -> 4x4 tiles
        int tr = tile >> 2, tc = tile & 3;
        const _Float16* arow = qh + (size_t)(blo + tr * 16 + m) * DIM_Q;
        const _Float16* brow = kh + (size_t)(blo + tc * 16 + m) * DIM_Q; // K contiguous
        v8f acc = {};
        #pragma unroll
        for (int k0 = 0; k0 < DIM_Q; k0 += 32) {
            v16h a = load_a_frag(arow, hi, k0);
            v16h bfr = load_b_frag(brow, hi, k0);
            acc = __builtin_amdgcn_wmma_f32_16x16x32_f16(false, a, false, bfr,
                                                         (short)0, acc, false, false);
        }
        float* outg = Ablk + (size_t)g * 4096;
        #pragma unroll
        for (int r = 0; r < 8; ++r)
            outg[(tr * 16 + hi * 8 + r) * 64 + tc * 16 + m] = acc[r];
    }
}

// ---------------------------------------------------------------------------
// K7: per-row streaming softmax stats over full N columns (bandwidth-bound).
// b (268MB) exceeds the 192MB L2 and is touched once -> non-temporal loads
// keep the stream from evicting the reusable working set (edge_attr, CSR,
// qh/kh/vhT, Ablk). Pair-columns are skipped via an LDS bitmap and folded in
// exactly from the CSR afterwards.
__global__ void k7_rowstats(const float* __restrict__ b,
                            const int* __restrict__ ptr, int nptr,
                            const float* __restrict__ Ablk,
                            const int* __restrict__ rowptr,
                            const int* __restrict__ colidx,
                            const float* __restrict__ val,
                            const float* __restrict__ attn_raw,
                            float* __restrict__ rowM,
                            float* __restrict__ rowS,
                            int N) {
    __shared__ unsigned bmap[256];                     // N=8192 bits
    __shared__ float rm[256], rs[256];
    int i = blockIdx.x, tid = threadIdx.x;

    // graph of row i: largest g with ptr[g] <= i
    int l = 0, r = nptr - 2;
    while (l < r) { int mid = (l + r + 1) >> 1; if (ptr[mid] <= i) l = mid; else r = mid - 1; }
    int g = l, blo = ptr[g], bhi = ptr[g + 1];

    float w0, w1; softmax_w(attn_raw, w0, w1);

    bmap[tid] = 0;
    __syncthreads();
    int ps = rowptr[i], pe = rowptr[i + 1];
    for (int p = ps + tid; p < pe; p += 256) {
        int d = colidx[p];
        atomicOr(&bmap[d >> 5], 1u << (d & 31));
    }
    __syncthreads();

    const float* brow = b + (size_t)i * N;
    const float* arow = Ablk + (size_t)g * 4096 + (size_t)(i - blo) * 64;
    float m = -3.0e38f, s = 0.0f;

    for (int j0 = tid * 4; j0 < N; j0 += 1024) {       // 256 thr x float4
        v4f b4 = __builtin_nontemporal_load((const v4f*)(brow + j0));
        #pragma unroll
        for (int q = 0; q < 4; ++q) {
            int j = j0 + q;
            if (bmap[j >> 5] & (1u << (j & 31))) continue;   // exact later
            float lg;
            if (j >= blo && j < bhi) lg = arow[j - blo] + w0 * b4[q];
            else                     lg = (w0 * b4[q]) * NEGF;
            online_update(m, s, lg);
        }
    }
    // exact logits at sparse-c columns
    for (int p = ps + tid; p < pe; p += 256) {
        int j = colidx[p];
        float lg = w0 * brow[j] + w1 * val[p];
        if (j >= blo && j < bhi) lg += arow[j - blo];
        else                     lg *= NEGF;
        online_update(m, s, lg);
    }

    rm[tid] = m; rs[tid] = s;
    __syncthreads();
    for (int st = 128; st > 0; st >>= 1) {
        if (tid < st) {
            float m2 = rm[tid + st], s2 = rs[tid + st];
            float mm = fmaxf(rm[tid], m2);
            rs[tid] = rs[tid] * __expf(rm[tid] - mm) + s2 * __expf(m2 - mm);
            rm[tid] = mm;
        }
        __syncthreads();
    }
    if (tid == 0) { rowM[i] = rm[0]; rowS[i] = rs[0]; }
}

// ---------------------------------------------------------------------------
// K8: per-graph attn_blk (f16, LDS) and out = attn_blk @ v_blk via WMMA
__global__ void k8_attnv(const float* __restrict__ b,
                         const int* __restrict__ ptr,
                         const float* __restrict__ Ablk,
                         const int* __restrict__ rowptr,
                         const int* __restrict__ colidx,
                         const float* __restrict__ val,
                         const float* __restrict__ rowM,
                         const float* __restrict__ rowS,
                         const _Float16* __restrict__ vhT,
                         const float* __restrict__ attn_raw,
                         float* __restrict__ out,
                         int N) {
    __shared__ float lg[4096];
    __shared__ alignas(32) _Float16 ah[4096];
    int g = blockIdx.x, tid = threadIdx.x;
    int blo = ptr[g], bhi = ptr[g + 1];
    float w0, w1; softmax_w(attn_raw, w0, w1);

    for (int idx = tid; idx < 4096; idx += 256) {
        int i = idx >> 6, j = idx & 63;
        lg[idx] = Ablk[(size_t)g * 4096 + idx] + w0 * b[(size_t)(blo + i) * N + blo + j];
    }
    __syncthreads();
    if (tid < 64) {                                    // in-block sparse c
        int i = blo + tid;
        for (int p = rowptr[i]; p < rowptr[i + 1]; ++p) {
            int d = colidx[p];
            if (d >= blo && d < bhi) lg[tid * 64 + (d - blo)] += w1 * val[p];
        }
    }
    __syncthreads();
    for (int idx = tid; idx < 4096; idx += 256) {
        int i = idx >> 6;
        float inv = 1.0f / rowS[blo + i];
        ah[idx] = (_Float16)(__expf(lg[idx] - rowM[blo + i]) * inv);
    }
    __syncthreads();

    int wave = tid >> 5, lane = tid & 31, m = lane & 15, hi = lane >> 4;
    #pragma unroll
    for (int t = 0; t < 2; ++t) {
        int tile = wave * 2 + t;
        int tr = tile >> 2, tc = tile & 3;
        const _Float16* arow = ah  + (size_t)(tr * 16 + m) * 64;          // LDS
        const _Float16* brow = vhT + (size_t)(tc * 16 + m) * N + blo;     // K contiguous
        v8f acc = {};
        #pragma unroll
        for (int k0 = 0; k0 < 64; k0 += 32) {
            v16h a = load_a_frag(arow, hi, k0);
            v16h bfr = load_b_frag(brow, hi, k0);
            acc = __builtin_amdgcn_wmma_f32_16x16x32_f16(false, a, false, bfr,
                                                         (short)0, acc, false, false);
        }
        #pragma unroll
        for (int r = 0; r < 8; ++r)
            out[(size_t)(blo + tr * 16 + hi * 8 + r) * DIM_Q + tc * 16 + m] = acc[r];
    }
}

// ---------------------------------------------------------------------------
extern "C" void kernel_launch(void* const* d_in, const int* in_sizes, int n_in,
                              void* d_out, int out_size, void* d_ws, size_t ws_size,
                              hipStream_t stream) {
    (void)n_in; (void)out_size; (void)ws_size;
    const float* x          = (const float*)d_in[0];
    const float* edge_attr  = (const float*)d_in[1];
    const float* b          = (const float*)d_in[2];
    const int*   src        = (const int*)  d_in[3];
    const int*   dst        = (const int*)  d_in[4];
    const int*   path       = (const int*)  d_in[5];
    const int*   ptr        = (const int*)  d_in[6];
    const float* Wq         = (const float*)d_in[7];
    const float* bq         = (const float*)d_in[8];
    const float* Wk         = (const float*)d_in[9];
    const float* bk         = (const float*)d_in[10];
    const float* Wv         = (const float*)d_in[11];
    const float* bv         = (const float*)d_in[12];
    const float* edge_vec   = (const float*)d_in[13];
    const float* attn_raw   = (const float*)d_in[14];
    float* out = (float*)d_out;

    const int N    = in_sizes[0] / DIM_IN;   // 8192
    const int P    = in_sizes[3];            // 400000
    const int nptr = in_sizes[6];            // 129
    const int NG   = nptr - 1;               // 128

    // workspace bump allocator (256B aligned)
    char* base = (char*)d_ws;
    size_t off = 0;
    auto alloc = [&](size_t bytes) -> void* {
        void* p = base + off;
        off = (off + bytes + 255) & ~(size_t)255;
        return p;
    };
    _Float16* xh      = (_Float16*)alloc((size_t)N * DIM_IN * 2);
    _Float16* WhT     = (_Float16*)alloc((size_t)QKV_COLS * DIM_IN * 2);
    _Float16* qh      = (_Float16*)alloc((size_t)N * DIM_Q * 2);
    _Float16* kh      = (_Float16*)alloc((size_t)N * DIM_Q * 2);
    _Float16* vhT     = (_Float16*)alloc((size_t)N * DIM_Q * 2);
    float*    Ablk    = (float*)   alloc((size_t)NG * 64 * 64 * 4);
    float*    rowMv   = (float*)   alloc((size_t)N * 4);
    float*    rowSv   = (float*)   alloc((size_t)N * 4);
    int*      rowcnt  = (int*)     alloc((size_t)N * 4);
    int*      rowfill = (int*)     alloc((size_t)N * 4);
    int*      rowptrW = (int*)     alloc((size_t)(N + 1) * 4);
    int*      colidx  = (int*)     alloc((size_t)P * 4);
    float*    valW    = (float*)   alloc((size_t)P * 4);

    // K1 convert + init
    hipLaunchKernelGGL(k1_convert, dim3(512), dim3(256), 0, stream,
                       x, Wq, Wk, Wv, xh, WhT, rowcnt, rowfill, N);
    // K2 QKV (WMMA)
    int nTiles = (N / 16) * NCT;
    hipLaunchKernelGGL(k2_qkv, dim3((nTiles + 7) / 8), dim3(256), 0, stream,
                       xh, WhT, bq, bk, bv, qh, kh, vhT, N, nTiles);
    // K3-K5 CSR of sparse c
    hipLaunchKernelGGL(k3_count, dim3((P + 255) / 256), dim3(256), 0, stream,
                       src, rowcnt, P);
    hipLaunchKernelGGL(k4_scan, dim3(1), dim3(256), 0, stream, rowcnt, rowptrW, N);
    hipLaunchKernelGGL(k5_dots_fill, dim3((P + 255) / 256), dim3(256), 0, stream,
                       edge_attr, src, dst, path, edge_vec, rowptrW, rowfill,
                       colidx, valW, P);
    // K6 block-diagonal qk^T (WMMA)
    hipLaunchKernelGGL(k6_ablk, dim3(NG), dim3(256), 0, stream, qh, kh, ptr, Ablk);
    // K7 full-row softmax stats (streams all of b once: the bandwidth roofline)
    hipLaunchKernelGGL(k7_rowstats, dim3(N), dim3(256), 0, stream,
                       b, ptr, nptr, Ablk, rowptrW, colidx, valW, attn_raw,
                       rowMv, rowSv, N);
    // K8 in-block attention * v (WMMA)
    hipLaunchKernelGGL(k8_attnv, dim3(NG), dim3(256), 0, stream,
                       b, ptr, Ablk, rowptrW, colidx, valW, rowMv, rowSv, vhT,
                       attn_raw, out, N);
}
